// CrossAttentionModel_53841710023091
// MI455X (gfx1250) — matrix operand order repaired and checked by reading, not compiled, not run on hardware
//
#include <hip/hip_runtime.h>
#include <cmath>

// ---------------------------------------------------------------------------
// CrossAttentionModel on MI455X (gfx1250, wave32, WMMA)
//
// Encoder GEMM (dominant, ~2.7 GFLOP): rows 0..383 = aud (b*3+c),
// rows 384..511 = unique vis rows (v is channel-replicated in the reference,
// so its GEMM is done once per batch, not 3x).
//   out[r,n] = sum_k A[r,k] * W[n,k]
// bf16 WMMA 16x16x32, f32 accumulate, split-K=20 with f32 atomics.
// LDS tiles are stored in *fragment layout* (16 contiguous bf16 per lane)
// so fragment loads are 2x ds_load_b128 per matrix.
// ---------------------------------------------------------------------------

typedef __attribute__((ext_vector_type(16))) __bf16 v16bf;
typedef __attribute__((ext_vector_type(8)))  float  v8f;

#define B_BATCH 128
#define HH 512
#define WW 640
#define KDIM 20480
#define DENC 128
#define DATT 32
#define M_AUD 384
#define M_TOT 512
#define KSPLIT 20
#define KSTEPS2 16      // 16 iters x 64 K x 20 chunks = 20480

// ISA 7.12.2 16-bit A 16x32 layout, forward map: element e -> K
__device__ __forceinline__ int frag_ka(int lane, int e) {
  return e + ((e >= 8) ? 8 : 0) + ((lane >= 16) ? 8 : 0);
}
// B 32x16 as staged here (A too): lanes 0-15 K=0..15, lanes 16-31 K=16..31

// --------------------------- bias init (accumulator seed) -------------------
__global__ __launch_bounds__(256) void k_init_bias(const float* __restrict__ b1,
                                                   const float* __restrict__ b2,
                                                   float* __restrict__ aud_buf,
                                                   float* __restrict__ vis_buf) {
  int i = blockIdx.x * blockDim.x + threadIdx.x;            // 65536 total
  if (i < M_AUD * DENC) aud_buf[i] = b1[i & (DENC - 1)];
  else                  vis_buf[i - M_AUD * DENC] = b2[i & (DENC - 1)];
}

// --------------------------- encoder GEMM (WMMA) ----------------------------
__global__ __launch_bounds__(256) void k_enc_gemm(
    const float* __restrict__ f1, const float* __restrict__ f2,
    const float* __restrict__ W1, const float* __restrict__ W2,
    float* __restrict__ aud_out, float* __restrict__ vis_out)
{
  // Fragment-layout LDS: [sub-kstep][lane][element], 32 B contiguous per lane.
  __shared__ alignas(32) __bf16 AsF[2][32][16];       // 2 KB
  __shared__ alignas(32) __bf16 BsF[2][8][32][16];    // 16 KB

  const int tid  = threadIdx.x;
  const int lane = tid & 31;
  const int wave = tid >> 5;          // 8 waves -> one 16-wide N tile each
  const int m0   = blockIdx.x * 16;   // 32 M tiles (24 aud + 8 vis)
  const bool isVis = (m0 >= M_AUD);
  const float* __restrict__ Wsel = isVis ? W2 : W1;
  const int kbase = blockIdx.y * (KSTEPS2 * 64);

  v8f acc = {};

  for (int ks = 0; ks < KSTEPS2; ++ks) {
    const int k0 = kbase + ks * 64;

    // ---- Stage A (16 rows x 64 K, strided 4x4-downsample gather, NT hint).
    // Inverse frag map: (row,kk) -> lane = row + 16*(kk>=16),
    //                              e    = kk&15
    #pragma unroll
    for (int j = 0; j < 4; ++j) {
      int idx = tid + j * 256;              // 1024 elements
      int sub = idx >> 9;
      int rw  = (idx >> 5) & 15;
      int kk  = idx & 31;
      int gk  = k0 + sub * 32 + kk;
      int hp  = gk / 160, wp = gk - hp * 160;
      int r   = m0 + rw;
      const float* src = (r < M_AUD)
          ? &f1[((size_t)r * HH + 4 * hp) * WW + 4 * wp]
          : &f2[((size_t)(r - M_AUD) * HH + 4 * hp) * WW + 4 * wp];
      float val = __builtin_nontemporal_load(src);
      AsF[sub][rw + 16 * (kk >= 16)][kk & 15] = (__bf16)val;
    }

    // ---- Stage B (64 K x 128 N from W[n][k], coalesced float4, L2-hot).
    // Inverse frag map: (kk,n) -> lane = (n&15) + 16*(kk>=16), e = kk&15
    #pragma unroll
    for (int j = 0; j < 8; ++j) {
      int idx = tid + j * 256;              // 2048 float4 loads
      int n   = idx >> 4;
      int rem = idx & 15;
      int sub = rem >> 3;
      int kq  = (rem & 7) * 4;
      const float4 w4 = *(const float4*)&Wsel[(size_t)n * KDIM + k0 + sub * 32 + kq];
      int nt = n >> 4, nl = n & 15;
      int lhi = 16 * (kq >= 16);            // kq multiple of 4: group shares half
      __bf16* dst = &BsF[sub][nt][nl + lhi][kq & 15];
      dst[0] = (__bf16)w4.x;
      dst[1] = (__bf16)w4.y;
      dst[2] = (__bf16)w4.z;
      dst[3] = (__bf16)w4.w;
    }
    __syncthreads();

    #pragma unroll
    for (int sub = 0; sub < 2; ++sub) {
      v16bf a = *(const v16bf*)&AsF[sub][lane][0];          // 2x ds_load_b128
      v16bf b = *(const v16bf*)&BsF[sub][wave][lane][0];    // 2x ds_load_b128
      acc = __builtin_amdgcn_wmma_f32_16x16x32_bf16(false, a, false, b,
                                                    (short)0, acc, false, false);
    }
    __syncthreads();
  }

  // ---- Split-K epilogue: uniform destination select, f32 atomics.
  float* __restrict__ dst = isVis ? vis_out : aud_out;
  const int rb = isVis ? (m0 - M_AUD) : m0;
  const int nn = wave * 16 + (lane & 15);
  const int mo = (lane >= 16) ? 8 : 0;
  float* base = &dst[(size_t)(rb + mo) * DENC + nn];
  #pragma unroll
  for (int v = 0; v < 8; ++v) atomicAdd(&base[(size_t)v * DENC], acc[v]);
}

// --------------------------- fused attention tail ---------------------------
__global__ __launch_bounds__(256) void k_attn(
    const float* __restrict__ aud_buf, const float* __restrict__ vis_buf,
    const float* __restrict__ Aa,  const float* __restrict__ Av,
    const float* __restrict__ Wa,  const float* __restrict__ Wv,
    const float* __restrict__ Wca, const float* __restrict__ Wcv,
    const float* __restrict__ Wha, const float* __restrict__ Whv,
    float* __restrict__ out)
{
  __shared__ float s_aud[3][128];
  __shared__ float s_vis[128];
  __shared__ float s_Aav[3][256];
  __shared__ float s_Vsum[256];
  __shared__ float s_Ha[128][DATT + 1];
  __shared__ float s_Hv[128][DATT + 1];

  const int b    = blockIdx.x;
  const int tid  = threadIdx.x;
  const int lane = tid & 31;
  const int wave = tid >> 5;

  for (int idx = tid; idx < 512; idx += 256) {
    if (idx < 384) s_aud[idx >> 7][idx & 127] = aud_buf[(size_t)(b * 3) * 128 + idx];
    else           s_vis[idx - 384]           = vis_buf[(size_t)b * 128 + (idx - 384)];
  }
  __syncthreads();

  { // Aav = Aa @ av ; Vsum[m] = sum_i (Av @ av)[i][m]  (vis rows identical)
    int m = tid;
    float av0 = (m < 128) ? s_aud[0][m] : s_vis[m - 128];
    float av1 = (m < 128) ? s_aud[1][m] : s_vis[m - 128];
    float av2 = (m < 128) ? s_aud[2][m] : s_vis[m - 128];
    float vs = 0.f;
    #pragma unroll
    for (int i = 0; i < 3; ++i) {
      s_Aav[i][m] = Aa[i * 3 + 0] * av0 + Aa[i * 3 + 1] * av1 + Aa[i * 3 + 2] * av2;
      vs         += Av[i * 3 + 0] * av0 + Av[i * 3 + 1] * av1 + Av[i * 3 + 2] * av2;
    }
    s_Vsum[m] = vs;
  }
  __syncthreads();

  const float scale = 0.0625f;            // 1/sqrt(256)
  const int k0w  = wave * 16;
  const int krow = k0w + (lane & 15);
  const int h    = lane & 15;
  const float a0 = s_aud[0][krow], a1 = s_aud[1][krow], a2 = s_aud[2][krow];
  const float vk = s_vis[krow];

  // H = relu( att(128x256) @ Wc^T(256x32) + lin ), tanh computed in-fragment.
  v8f hA0 = {}, hA1 = {}, hV0 = {}, hV1 = {};
  for (int kc = 0; kc < 8; ++kc) {
    v16bf aA, aV;
    #pragma unroll
    for (int e = 0; e < 16; ++e) {
      int m = kc * 32 + e + ((lane >= 16) ? 16 : 0);
      float sa = scale * (a0 * s_Aav[0][m] + a1 * s_Aav[1][m] + a2 * s_Aav[2][m]);
      aA[e] = (__bf16)tanhf(sa);
      aV[e] = (__bf16)tanhf(scale * vk * s_Vsum[m]);
    }
    // B-fragment: element e -> m = kc*32 + e + 16*(lane>=16): 16 consecutive
    // floats per lane -> 4x float4 (global_load_b128) per fragment.
    const int mb = kc * 32 + ((lane >= 16) ? 16 : 0);
    v16bf bA0, bA1, bV0, bV1;
    #pragma unroll
    for (int q = 0; q < 4; ++q) {
      float4 wa0 = *(const float4*)&Wca[(size_t)h * 256 + mb + 4 * q];
      float4 wa1 = *(const float4*)&Wca[(size_t)(h + 16) * 256 + mb + 4 * q];
      float4 wv0 = *(const float4*)&Wcv[(size_t)h * 256 + mb + 4 * q];
      float4 wv1 = *(const float4*)&Wcv[(size_t)(h + 16) * 256 + mb + 4 * q];
      bA0[4*q+0] = (__bf16)wa0.x; bA0[4*q+1] = (__bf16)wa0.y;
      bA0[4*q+2] = (__bf16)wa0.z; bA0[4*q+3] = (__bf16)wa0.w;
      bA1[4*q+0] = (__bf16)wa1.x; bA1[4*q+1] = (__bf16)wa1.y;
      bA1[4*q+2] = (__bf16)wa1.z; bA1[4*q+3] = (__bf16)wa1.w;
      bV0[4*q+0] = (__bf16)wv0.x; bV0[4*q+1] = (__bf16)wv0.y;
      bV0[4*q+2] = (__bf16)wv0.z; bV0[4*q+3] = (__bf16)wv0.w;
      bV1[4*q+0] = (__bf16)wv1.x; bV1[4*q+1] = (__bf16)wv1.y;
      bV1[4*q+2] = (__bf16)wv1.z; bV1[4*q+3] = (__bf16)wv1.w;
    }
    hA0 = __builtin_amdgcn_wmma_f32_16x16x32_bf16(false, aA, false, bA0, (short)0, hA0, false, false);
    hA1 = __builtin_amdgcn_wmma_f32_16x16x32_bf16(false, aA, false, bA1, (short)0, hA1, false, false);
    hV0 = __builtin_amdgcn_wmma_f32_16x16x32_bf16(false, aV, false, bV0, (short)0, hV0, false, false);
    hV1 = __builtin_amdgcn_wmma_f32_16x16x32_bf16(false, aV, false, bV1, (short)0, hV1, false, false);
  }

  { // + Wa/Wv linear term, relu, stash H in LDS
    int mo = (lane >= 16) ? 8 : 0;
    float wa0  = Wa[h * 3 + 0],        wa1  = Wa[h * 3 + 1],        wa2  = Wa[h * 3 + 2];
    float wb0  = Wa[(h + 16) * 3 + 0], wb1  = Wa[(h + 16) * 3 + 1], wb2  = Wa[(h + 16) * 3 + 2];
    float wvs  = Wv[h * 3 + 0] + Wv[h * 3 + 1] + Wv[h * 3 + 2];
    float wvsb = Wv[(h + 16) * 3 + 0] + Wv[(h + 16) * 3 + 1] + Wv[(h + 16) * 3 + 2];
    #pragma unroll
    for (int v = 0; v < 8; ++v) {
      int k = k0w + v + mo;
      float A0 = s_aud[0][k], A1 = s_aud[1][k], A2 = s_aud[2][k];
      s_Ha[k][h]      = fmaxf(hA0[v] + A0 * wa0 + A1 * wa1 + A2 * wa2, 0.f);
      s_Ha[k][h + 16] = fmaxf(hA1[v] + A0 * wb0 + A1 * wb1 + A2 * wb2, 0.f);
      s_Hv[k][h]      = fmaxf(hV0[v] + s_vis[k] * wvs, 0.f);
      s_Hv[k][h + 16] = fmaxf(hV1[v] + s_vis[k] * wvsb, 0.f);
    }
  }
  __syncthreads();

  { // out[b][c][m] = H @ Wh^T + residual
    int m = tid;
    #pragma unroll
    for (int c = 0; c < 3; ++c) {
      float acc;
      if (m < 128) {
        acc = s_aud[c][m];
        #pragma unroll
        for (int hh = 0; hh < DATT; ++hh) acc += s_Ha[m][hh] * Wha[c * DATT + hh];
      } else {
        int k = m - 128;
        acc = s_vis[k];
        #pragma unroll
        for (int hh = 0; hh < DATT; ++hh) acc += s_Hv[k][hh] * Whv[c * DATT + hh];
      }
      out[((size_t)b * 3 + c) * 256 + m] = acc;
    }
  }
}

// ---------------------------------------------------------------------------
extern "C" void kernel_launch(void* const* d_in, const int* in_sizes, int n_in,
                              void* d_out, int out_size, void* d_ws, size_t ws_size,
                              hipStream_t stream) {
  const float* f1  = (const float*)d_in[0];
  const float* f2  = (const float*)d_in[1];
  const float* W1  = (const float*)d_in[2];
  const float* b1  = (const float*)d_in[3];
  const float* W2  = (const float*)d_in[4];
  const float* b2  = (const float*)d_in[5];
  const float* Aa  = (const float*)d_in[6];
  const float* Av  = (const float*)d_in[7];
  const float* Wa  = (const float*)d_in[8];
  const float* Wv  = (const float*)d_in[9];
  const float* Wca = (const float*)d_in[10];
  const float* Wcv = (const float*)d_in[11];
  const float* Wha = (const float*)d_in[12];
  const float* Whv = (const float*)d_in[13];
  float* out = (float*)d_out;

  float* aud_buf = (float*)d_ws;               // 384*128 f32
  float* vis_buf = aud_buf + M_AUD * DENC;     // 128*128 f32

  k_init_bias<<<256, 256, 0, stream>>>(b1, b2, aud_buf, vis_buf);

  dim3 g1(M_TOT / 16, KSPLIT);                 // 32 M-tiles x 20 K-chunks
  k_enc_gemm<<<g1, 256, 0, stream>>>(f1, f2, W1, W2, aud_buf, vis_buf);

  k_attn<<<B_BATCH, 256, 0, stream>>>(aud_buf, vis_buf, Aa, Av, Wa, Wv,
                                      Wca, Wcv, Wha, Whv, out);
}
